// GCN_Net3_60859686584749
// MI455X (gfx1250) — matrix operand order repaired and checked by reading, not compiled, not run on hardware
//
#include <hip/hip_runtime.h>
#include <hip/hip_bf16.h>

typedef __attribute__((ext_vector_type(16))) _Float16 v16h;
typedef __attribute__((ext_vector_type(8)))  float    v8f;

#define TPB 256

// ---------------------------------------------------------------- edge prep
__global__ void init_edges_k(const long long* __restrict__ src64,
                             const long long* __restrict__ dst64,
                             int* __restrict__ esrc, int* __restrict__ edst,
                             float* __restrict__ ew, int E) {
    int e = blockIdx.x * blockDim.x + threadIdx.x;
    if (e < E) { esrc[e] = (int)src64[e]; edst[e] = (int)dst64[e]; ew[e] = 1.0f; }
}

__global__ void deg_k(const int* __restrict__ edst, const float* __restrict__ ew,
                      float* __restrict__ deg, int E) {
    int e = blockIdx.x * blockDim.x + threadIdx.x;
    if (e < E) { float w = ew[e]; if (w != 0.0f) atomicAdd(&deg[edst[e]], w); }
}

__global__ void dis_k(float* __restrict__ d, int N) {   // deg -> deg^{-1/2} in place
    int i = blockIdx.x * blockDim.x + threadIdx.x;
    if (i < N) d[i] = rsqrtf(d[i] + 1.0f);
}

// layer-1 scalar aggregation: s[dst] += dis[src]*dis[dst]*w*x[src]
__global__ void agg_scalar_k(const int* __restrict__ esrc, const int* __restrict__ edst,
                             const float* __restrict__ ew, const float* __restrict__ dis,
                             const float* __restrict__ x, float* __restrict__ sagg, int E) {
    int e = blockIdx.x * blockDim.x + threadIdx.x;
    if (e < E) {
        float w = ew[e];
        if (w != 0.0f) {
            int s = esrc[e], d = edst[e];
            atomicAdd(&sagg[d], dis[s] * dis[d] * w * x[s]);
        }
    }
}

// layer-1 output: out[i][c] = (s[i] + dis[i]^2 * x[i]) * W1[c] + b1[c]
__global__ void layer1_out_k(const float* __restrict__ x, const float* __restrict__ sagg,
                             const float* __restrict__ dis, const float* __restrict__ W1,
                             const float* __restrict__ b1, float* __restrict__ out, int N) {
    int t = blockIdx.x * blockDim.x + threadIdx.x;
    if (t < N * 64) {
        int i = t >> 6, c = t & 63;
        float di = dis[i];
        out[t] = (sagg[i] + di * di * x[i]) * W1[c] + b1[c];
    }
}

// ---------------------------------------------------------------- message passing (C=64)
__global__ void msg_k(const int* __restrict__ esrc, const int* __restrict__ edst,
                      const float* __restrict__ ew, const float* __restrict__ dis,
                      const float* __restrict__ xw, float* __restrict__ msg, int E) {
    int t = blockIdx.x * blockDim.x + threadIdx.x;   // E*8 threads, 8 channels each
    int e = t >> 3;
    if (e < E) {
        float w = ew[e];
        if (w != 0.0f) {
            int s = esrc[e], d = edst[e];
            float coef = dis[s] * dis[d] * w;
            int c0 = (t & 7) << 3;
            const float* xs = xw + (size_t)s * 64 + c0;
            float* md = msg + (size_t)d * 64 + c0;
#pragma unroll
            for (int k = 0; k < 8; ++k) atomicAdd(&md[k], coef * xs[k]);
        }
    }
}

__global__ void combine_k(float* __restrict__ msg, const float* __restrict__ xw,
                          const float* __restrict__ dis, const float* __restrict__ b, int N) {
    int t = blockIdx.x * blockDim.x + threadIdx.x;
    if (t < N * 64) {
        int i = t >> 6, c = t & 63;
        float di = dis[i];
        msg[t] = msg[t] + di * di * xw[t] + b[c];
    }
}

// ---------------------------------------------------------------- pooling
__global__ void pool_nodes_k(const float* __restrict__ in, float* __restrict__ out,
                             int N, int newN, int stride) {
    int t = blockIdx.x * blockDim.x + threadIdx.x;
    if (t < newN * 64) {
        int i = t >> 6, c = t & 63;
        float sum = 0.0f;
        for (int j = 0; j < stride; ++j) {
            int r = i * stride + j;
            if (r < N) sum += in[(size_t)r * 64 + c];
        }
        out[t] = sum / (float)stride;
    }
}

__global__ void pool_edges_k(int* __restrict__ esrc, int* __restrict__ edst,
                             float* __restrict__ ew, int E, int stride, int newN) {
    int e = blockIdx.x * blockDim.x + threadIdx.x;
    if (e < E) {
        int s = esrc[e], d = edst[e];
        bool keep = ((s % stride) == 0) && ((d % stride) == 0);
        if (!keep) ew[e] = 0.0f;
        int ns = s / stride; if (ns > newN - 1) ns = newN - 1;
        int nd = d / stride; if (nd > newN - 1) nd = newN - 1;
        esrc[e] = ns; edst[e] = nd;
    }
}

// ---------------------------------------------------------------- batchnorm
__global__ void bn_stats_k(const float* __restrict__ xf, float* __restrict__ acc, int N) {
    int g = blockIdx.x * blockDim.x + threadIdx.x;   // 64*256 threads
    int c = g & 63, r0 = g >> 6;
    float s = 0.0f, ss = 0.0f;
    for (int r = r0; r < N; r += 256) {
        float v = xf[(size_t)r * 64 + c];
        s += v; ss += v * v;
    }
    atomicAdd(&acc[c], s);
    atomicAdd(&acc[64 + c], ss);
}

__global__ void bn_final_k(const float* __restrict__ acc, const float* __restrict__ g,
                           const float* __restrict__ be, float* __restrict__ sc, int N) {
    int c = threadIdx.x;
    if (c < 64) {
        float mu  = acc[c] / (float)N;
        float var = acc[64 + c] / (float)N - mu * mu;
        float a = g[c] * rsqrtf(var + 1e-5f);
        sc[c] = a;
        sc[64 + c] = be[c] - mu * a;
    }
}

// ---------------------------------------------------------------- WMMA operand packing
// A blob: chunk (tm,kc) holds 32 lanes x 16 f16 in register order:
//   row = tm*16 + lane%16, k = kc*32 + (i<8?0:16) + (lane/16)*8 + (i&7)
// Optionally fuses BN scale/shift + ReLU (sc != nullptr).
__global__ void pack_A_k(const float* __restrict__ src, const float* __restrict__ sc,
                         _Float16* __restrict__ dst, int M, int K, int kChunks, int tilesM) {
    int t = blockIdx.x * blockDim.x + threadIdx.x;
    if (t >= tilesM * kChunks * 512) return;
    int i    = t & 15;
    int lane = (t >> 4) & 31;
    int idx  = t >> 9;
    int kc   = idx % kChunks;
    int tm   = idx / kChunks;
    int grp = lane >> 4, mr = lane & 15;
    int row = tm * 16 + mr;
    int k   = kc * 32 + ((i < 8) ? 0 : 16) + grp * 8 + (i & 7);
    float v = 0.0f;
    if (row < M) {
        v = src[(size_t)row * K + k];
        if (sc) v = fmaxf(v * sc[k] + sc[64 + k], 0.0f);
    }
    dst[t] = (_Float16)v;
}

// B blob: chunk (kc,tn) at offset (kc*tilesN+tn)*512:
//   k = kc*32 + (i<8?0:16) + (lane/16)*8 + (i&7), col = tn*16 + lane%16
__global__ void pack_B_k(const float* __restrict__ B, _Float16* __restrict__ dst,
                         int K, int N, int kChunks, int tilesN) {
    int t = blockIdx.x * blockDim.x + threadIdx.x;
    if (t >= kChunks * tilesN * 512) return;
    int i    = t & 15;
    int lane = (t >> 4) & 31;
    int idx  = t >> 9;
    int tn   = idx % tilesN;
    int kc   = idx / tilesN;
    int grp = lane >> 4, nr = lane & 15;
    int k   = kc * 32 + ((i < 8) ? 0 : 16) + grp * 8 + (i & 7);
    int col = tn * 16 + nr;
    float v = (col < N) ? B[(size_t)k * N + col] : 0.0f;
    dst[t] = (_Float16)v;
}

// ---------------------------------------------------------------- WMMA GEMM
// C(MxN) = Apacked @ Bpacked [+bias] [relu]; one wave per 16x16 tile.
// Operands are pre-swizzled: each lane's v16h is 32 contiguous bytes (2x b128 loads).
__global__ void gemm_wmma_k(const _Float16* __restrict__ Ap, const _Float16* __restrict__ Bp,
                            const float* __restrict__ bias, float* __restrict__ C,
                            int M, int kChunks, int N, int ldc, int relu) {
    int tilesN = (N + 15) >> 4;
    int tilesM = (M + 15) >> 4;
    int wave = blockIdx.x * (blockDim.x >> 5) + (threadIdx.x >> 5);
    if (wave >= tilesM * tilesN) return;
    int tm = wave / tilesN, tn = wave % tilesN;
    int lane = threadIdx.x & 31;

    const v16h* Av = (const v16h*)Ap;
    const v16h* Bv = (const v16h*)Bp;
    v8f acc = {};
    for (int c = 0; c < kChunks; ++c) {
        v16h a = Av[((size_t)tm * kChunks + c) * 32 + lane];
        v16h b = Bv[((size_t)c * tilesN + tn) * 32 + lane];
        acc = __builtin_amdgcn_wmma_f32_16x16x32_f16(false, a, false, b,
                                                     (short)0, acc, false, false);
    }
    int grp = lane >> 4, mr = lane & 15;
    int n = tn * 16 + mr;
    float bv = (bias != nullptr && n < N) ? bias[n] : 0.0f;
#pragma unroll
    for (int r = 0; r < 8; ++r) {
        int m = tm * 16 + r + grp * 8;
        if (m < M && n < N) {
            float v = acc[r] + bv;
            if (relu) v = fmaxf(v, 0.0f);
            C[(size_t)m * ldc + n] = v;
        }
    }
}

// ---------------------------------------------------------------- FC pad-row broadcast
__global__ void padrow_k(const float* __restrict__ fcb1, const float* __restrict__ fcW2,
                         const float* __restrict__ fcb2, float* __restrict__ pr) {
    int n = threadIdx.x;
    if (n < 10) {
        float sum = fcb2[n];
        for (int k = 0; k < 128; ++k) sum += fmaxf(fcb1[k], 0.0f) * fcW2[k * 10 + n];
        pr[n] = sum;
    }
}

__global__ void fill_pad_k(const float* __restrict__ pr, float* __restrict__ out,
                           int startRow, int totalRows) {
    int t = blockIdx.x * blockDim.x + threadIdx.x;
    int cnt = (totalRows - startRow) * 10;
    if (t < cnt) {
        int row = startRow + t / 10;
        out[(size_t)row * 10 + (t % 10)] = pr[t % 10];
    }
}

// ---------------------------------------------------------------- launch
static inline int cdiv(int a, int b) { return (a + b - 1) / b; }

extern "C" void kernel_launch(void* const* d_in, const int* in_sizes, int n_in,
                              void* d_out, int out_size, void* d_ws, size_t ws_size,
                              hipStream_t stream) {
    const float* x    = (const float*)d_in[0];
    const float* W[4]  = {(const float*)d_in[1],  (const float*)d_in[5],
                          (const float*)d_in[9],  (const float*)d_in[13]};
    const float* bb[4] = {(const float*)d_in[2],  (const float*)d_in[6],
                          (const float*)d_in[10], (const float*)d_in[14]};
    const float* gg[4] = {(const float*)d_in[3],  (const float*)d_in[7],
                          (const float*)d_in[11], (const float*)d_in[15]};
    const float* be[4] = {(const float*)d_in[4],  (const float*)d_in[8],
                          (const float*)d_in[12], (const float*)d_in[16]};
    const float* fcW1 = (const float*)d_in[17];
    const float* fcb1 = (const float*)d_in[18];
    const float* fcW2 = (const float*)d_in[19];
    const float* fcb2 = (const float*)d_in[20];
    const long long* src64 = (const long long*)d_in[21];
    const long long* dst64 = (const long long*)d_in[22];
    float* out = (float*)d_out;

    const int N0 = in_sizes[0];      // 100000
    const int E  = in_sizes[21];     // 1600000

    // ---- workspace layout (bump allocator, 256B aligned) ----
    char* p = (char*)d_ws;
    auto alloc = [&](size_t bytes) -> void* {
        void* r = p; p += (bytes + 255) & ~(size_t)255; return r;
    };
    float*     bufA  = (float*)alloc((size_t)N0 * 64 * 4);             // pre-pool out / msg acc
    float*     bufX  = (float*)alloc((size_t)((N0 + 1) / 2) * 64 * 4); // pooled f32 (BN input)
    float*     bufXW = (float*)alloc((size_t)((N0 + 1) / 2) * 64 * 4); // xw for layers 2-4
    float*     dis   = (float*)alloc((size_t)N0 * 4);
    float*     sagg  = (float*)alloc((size_t)N0 * 4);
    int*       esrc  = (int*)  alloc((size_t)E * 4);
    int*       edst  = (int*)  alloc((size_t)E * 4);
    float*     eww   = (float*)alloc((size_t)E * 4);
    float*     hfc   = (float*)alloc((size_t)4096 * 128 * 4);          // fc1 activations
    _Float16*  packA = (_Float16*)alloc((size_t)8 * 1024 * 1024);      // swizzled A panels
    _Float16*  packB = (_Float16*)alloc((size_t)64 * 1024);            // swizzled B
    float*     bnacc = (float*)alloc(128 * 4);
    float*     bnsc  = (float*)alloc(128 * 4);
    float*     prow  = (float*)alloc(64);

    auto gemm = [&](const _Float16* Ap, const float* B, const float* bias, float* C,
                    int M, int K, int N, int ldc, int relu) {
        int kChunks = K / 32;
        int tilesM = cdiv(M, 16), tilesN = cdiv(N, 16);
        pack_B_k<<<cdiv(kChunks * tilesN * 512, TPB), TPB, 0, stream>>>(
            B, packB, K, N, kChunks, tilesN);
        gemm_wmma_k<<<cdiv(tilesM * tilesN, 8), 256, 0, stream>>>(
            Ap, packB, bias, C, M, kChunks, N, ldc, relu);
    };

    // ================= Layer 1 (rank-1: C_in = 1) =================
    init_edges_k<<<cdiv(E, TPB), TPB, 0, stream>>>(src64, dst64, esrc, edst, eww, E);
    hipMemsetAsync(dis, 0, (size_t)N0 * 4, stream);
    hipMemsetAsync(sagg, 0, (size_t)N0 * 4, stream);
    deg_k<<<cdiv(E, TPB), TPB, 0, stream>>>(edst, eww, dis, E);
    dis_k<<<cdiv(N0, TPB), TPB, 0, stream>>>(dis, N0);
    agg_scalar_k<<<cdiv(E, TPB), TPB, 0, stream>>>(esrc, edst, eww, dis, x, sagg, E);
    layer1_out_k<<<cdiv(N0 * 64, TPB), TPB, 0, stream>>>(x, sagg, dis, W[0], bb[0], bufA, N0);

    int Ncur = N0;
    const int strides[4] = {2, 2, 3, 3};
    // pool + bn(+pack) for layer 1
    {
        int s = strides[0];
        int newN = cdiv(Ncur, s);
        pool_nodes_k<<<cdiv(newN * 64, TPB), TPB, 0, stream>>>(bufA, bufX, Ncur, newN, s);
        pool_edges_k<<<cdiv(E, TPB), TPB, 0, stream>>>(esrc, edst, eww, E, s, newN);
        Ncur = newN;
        hipMemsetAsync(bnacc, 0, 128 * 4, stream);
        bn_stats_k<<<64, TPB, 0, stream>>>(bufX, bnacc, Ncur);
        bn_final_k<<<1, 64, 0, stream>>>(bnacc, gg[0], be[0], bnsc, Ncur);
        int tilesM = cdiv(Ncur, 16);
        pack_A_k<<<cdiv(tilesM * 2 * 512, TPB), TPB, 0, stream>>>(
            bufX, bnsc, packA, Ncur, 64, 2, tilesM);
    }

    // ================= Layers 2-4 (C = 64) =================
    for (int li = 1; li < 4; ++li) {
        // xw = h @ W  (WMMA on pre-swizzled operands)
        gemm(packA, W[li], nullptr, bufXW, Ncur, 64, 64, 64, 0);
        // degree + norm
        hipMemsetAsync(dis, 0, (size_t)Ncur * 4, stream);
        deg_k<<<cdiv(E, TPB), TPB, 0, stream>>>(edst, eww, dis, E);
        dis_k<<<cdiv(Ncur, TPB), TPB, 0, stream>>>(dis, Ncur);
        // message scatter
        hipMemsetAsync(bufA, 0, (size_t)Ncur * 64 * 4, stream);
        msg_k<<<cdiv(E * 8, TPB), TPB, 0, stream>>>(esrc, edst, eww, dis, bufXW, bufA, E);
        combine_k<<<cdiv(Ncur * 64, TPB), TPB, 0, stream>>>(bufA, bufXW, dis, bb[li], Ncur);
        // pool
        int s = strides[li];
        int newN = cdiv(Ncur, s);
        pool_nodes_k<<<cdiv(newN * 64, TPB), TPB, 0, stream>>>(bufA, bufX, Ncur, newN, s);
        pool_edges_k<<<cdiv(E, TPB), TPB, 0, stream>>>(esrc, edst, eww, E, s, newN);
        Ncur = newN;
        // bn + fused relu/pack
        hipMemsetAsync(bnacc, 0, 128 * 4, stream);
        bn_stats_k<<<64, TPB, 0, stream>>>(bufX, bnacc, Ncur);
        bn_final_k<<<1, 64, 0, stream>>>(bnacc, gg[li], be[li], bnsc, Ncur);
        int tilesM = cdiv(Ncur, 16);
        pack_A_k<<<cdiv(tilesM * 2 * 512, TPB), TPB, 0, stream>>>(
            bufX, bnsc, packA, Ncur, 64, 2, tilesM);
    }

    // ================= FC head =================
    // Real rows (Ncur = 2778): two WMMA GEMMs.
    gemm(packA, fcW1, fcb1, hfc, Ncur, 64, 128, 128, 1);   // relu(h @ fcW1 + fcb1)
    {
        int tilesM = cdiv(Ncur, 16);
        pack_A_k<<<cdiv(tilesM * 4 * 512, TPB), TPB, 0, stream>>>(
            hfc, nullptr, packA, Ncur, 128, 4, tilesM);
    }
    gemm(packA, fcW2, fcb2, out, Ncur, 128, 10, 10, 0);    // @ fcW2 + fcb2
    // Zero-padded rows are all identical: compute once, broadcast.
    padrow_k<<<1, 32, 0, stream>>>(fcb1, fcW2, fcb2, prow);
    int padCnt = (N0 - Ncur) * 10;
    fill_pad_k<<<cdiv(padCnt, TPB), TPB, 0, stream>>>(prow, out, Ncur, N0);
}
